// ReferenceBlock_83665962926538
// MI455X (gfx1250) — compile-verified, matrix operand order checked by reading, not run on hardware
//
#include <hip/hip_runtime.h>
#include <hip/hip_bf16.h>
#include <math.h>

// ---------------------------------------------------------------------------
// Gated DeltaNet block for MI455X (gfx1250, wave32, WMMA).
// GEMMs run on v_wmma_f32_16x16x32_f16 (f16 in, f32 accum) with a 64x64 wave
// tile (16 WMMAs per 8 operand-tile loads); the recurrent delta scan keeps
// the 128x256 per-head state fully register-resident.
// ---------------------------------------------------------------------------

#define T_LEN 4096
#define D_DIM 2048
#define H_N   16
#define DK_N  128
#define DV_N  256
#define KEY_DIM 2048   // H*DK
#define VAL_DIM 4096   // H*DV

typedef _Float16 h8   __attribute__((ext_vector_type(8)));
typedef _Float16 v16h __attribute__((ext_vector_type(16)));
typedef float    v8f  __attribute__((ext_vector_type(8)));

union V16U { v16h v; h8 h[2]; };

// ---------------------------------------------------------------------------
// WMMA GEMM:  C[M,N] (f32) = A[M,K] (f16, row-major) * B[N,K]^T (f16, row-major)
// Block = 256 threads = 8 waves (2x4), block tile 128x256, wave tile 64x64.
// Operand layout per CDNA5 ISA 7.12.2 (16-bit A 16x32): lane l holds row
// (l&15), K-chunks at base=(l>>4)*8 and base+16. B (K-major columns) loads
// with the identical pattern from W's row-major storage.
// ---------------------------------------------------------------------------
__device__ __forceinline__ v16h load_frag(const _Float16* __restrict__ p,
                                          int ld, int row0, int kk, int lane) {
  const int r = row0 + (lane & 15);
  const int b = kk + (lane >> 4) * 8;
  const _Float16* base = p + (size_t)r * ld + b;
  V16U u;
  u.h[0] = *(const h8*)(base);
  u.h[1] = *(const h8*)(base + 16);
  return u.v;
}

__global__ __launch_bounds__(256, 1) void gemm_f16_wmma(
    const _Float16* __restrict__ A, const _Float16* __restrict__ B,
    float* __restrict__ C, int N, int K) {
  const int lane = threadIdx.x & 31;
  const int wave = threadIdx.x >> 5;                  // 0..7
  const int row0 = blockIdx.y * 128 + (wave >> 2) * 64;
  const int col0 = blockIdx.x * 256 + (wave & 3) * 64;

  v8f acc[4][4];
#pragma unroll
  for (int i = 0; i < 4; ++i)
#pragma unroll
    for (int j = 0; j < 4; ++j) acc[i][j] = (v8f){};

  for (int kk = 0; kk < K; kk += 32) {
    v16h a[4], b[4];
#pragma unroll
    for (int i = 0; i < 4; ++i) a[i] = load_frag(A, K, row0 + 16 * i, kk, lane);
#pragma unroll
    for (int j = 0; j < 4; ++j) b[j] = load_frag(B, K, col0 + 16 * j, kk, lane);
#pragma unroll
    for (int i = 0; i < 4; ++i)
#pragma unroll
      for (int j = 0; j < 4; ++j)
        acc[i][j] = __builtin_amdgcn_wmma_f32_16x16x32_f16(
            false, a[i], false, b[j], (short)0, acc[i][j], false, false);
  }

  const int n  = lane & 15;
  const int mo = (lane >> 4) * 8;
#pragma unroll
  for (int i = 0; i < 4; ++i)
#pragma unroll
    for (int j = 0; j < 4; ++j)
#pragma unroll
      for (int v = 0; v < 8; ++v)
        C[(size_t)(row0 + 16 * i + mo + v) * N + col0 + 16 * j + n] = acc[i][j][v];
}

// ---------------------------------------------------------------------------
// Elementwise / reduction helpers
// ---------------------------------------------------------------------------
__global__ void cvt_f32_f16(const float* __restrict__ in, _Float16* __restrict__ out, int n) {
  int i = blockIdx.x * 256 + threadIdx.x;
  if (i < n) out[i] = (_Float16)in[i];
}

__global__ __launch_bounds__(256) void rmsnorm_cast_kernel(
    const float* __restrict__ x, const float* __restrict__ w,
    float* __restrict__ xn, _Float16* __restrict__ xn16) {
  const int t = blockIdx.x, tid = threadIdx.x;
  __shared__ float red[256];
  const float* xr = x + (size_t)t * D_DIM;
  float s = 0.f;
  for (int i = tid; i < D_DIM; i += 256) { float v = xr[i]; s += v * v; }
  red[tid] = s; __syncthreads();
  for (int st = 128; st > 0; st >>= 1) { if (tid < st) red[tid] += red[tid + st]; __syncthreads(); }
  const float rn = rsqrtf(red[0] / (float)D_DIM + 1e-5f);
  for (int i = tid; i < D_DIM; i += 256) {
    float v = xr[i] * rn * w[i];
    xn[(size_t)t * D_DIM + i]   = v;
    xn16[(size_t)t * D_DIM + i] = (_Float16)v;
  }
}

// depthwise causal conv (K=4) + SiLU + per-head L2 norm (* scale) for q/k
__global__ __launch_bounds__(128) void conv_norm_qk(
    const float* __restrict__ pre, const float* __restrict__ w,
    float* __restrict__ out, float scale) {
  const int bid = blockIdx.x;            // t*H + h
  const int t = bid >> 4;
  const int dk = threadIdx.x;
  const int c = (bid & 15) * DK_N + dk;  // channel in KEY_DIM
  float y = 0.f;
#pragma unroll
  for (int i = 0; i < 4; ++i) {
    int tt = t - 3 + i;
    if (tt >= 0) y += w[c * 4 + i] * pre[(size_t)tt * KEY_DIM + c];
  }
  y = y / (1.f + expf(-y));              // silu
  __shared__ float red[128];
  red[dk] = y * y; __syncthreads();
  for (int st = 64; st > 0; st >>= 1) { if (dk < st) red[dk] += red[dk + st]; __syncthreads(); }
  const float nrm = fmaxf(sqrtf(red[0]), 1e-12f);
  out[(size_t)bid * DK_N + dk] = y / nrm * scale;
}

// depthwise causal conv (K=4) + SiLU for v
__global__ __launch_bounds__(256) void conv_silu_v(
    const float* __restrict__ pre, const float* __restrict__ w, float* __restrict__ out) {
  const size_t gid = (size_t)blockIdx.x * 256 + threadIdx.x;
  const int t = (int)(gid >> 12);
  const int c = (int)(gid & (VAL_DIM - 1));
  float y = 0.f;
#pragma unroll
  for (int i = 0; i < 4; ++i) {
    int tt = t - 3 + i;
    if (tt >= 0) y += w[c * 4 + i] * pre[(size_t)tt * VAL_DIM + c];
  }
  out[gid] = y / (1.f + expf(-y));
}

// beta = sigmoid(xn @ b_proj.T); g = -exp(A_log)*softplus(xn @ a_proj.T + dt_bias)
__global__ __launch_bounds__(128) void betag_kernel(
    const float* __restrict__ xn, const float* __restrict__ a_proj,
    const float* __restrict__ b_proj, const float* __restrict__ A_log,
    const float* __restrict__ dt_bias, float* __restrict__ beta, float* __restrict__ g) {
  const int bid = blockIdx.x;            // t*H + h
  const int t = bid >> 4, h = bid & 15, tid = threadIdx.x;
  const float* xr = xn + (size_t)t * D_DIM;
  const float* ar = a_proj + (size_t)h * D_DIM;
  const float* br = b_proj + (size_t)h * D_DIM;
  float sa = 0.f, sb = 0.f;
  for (int i = tid; i < D_DIM; i += 128) {
    float xv = xr[i]; sa += xv * ar[i]; sb += xv * br[i];
  }
  __shared__ float ra[128], rb[128];
  ra[tid] = sa; rb[tid] = sb; __syncthreads();
  for (int st = 64; st > 0; st >>= 1) {
    if (tid < st) { ra[tid] += ra[tid + st]; rb[tid] += rb[tid + st]; }
    __syncthreads();
  }
  if (tid == 0) {
    beta[bid] = 1.f / (1.f + expf(-rb[0]));
    float da = ra[0] + dt_bias[h];
    float dt = (da > 20.f) ? da : log1pf(expf(da));
    g[bid] = -expf(A_log[h]) * dt;
  }
}

// ---------------------------------------------------------------------------
// Gated delta-rule scan: one block per head, thread dv owns state column
// h[:,dv] in 128 VGPRs (fully unrolled). k/q broadcast via double-buffered
// LDS, one barrier per step. kh uses decayed state: kt@(h*eg) = eg*(kt@h).
// ---------------------------------------------------------------------------
__global__ __launch_bounds__(256) void delta_scan_kernel(
    const float* __restrict__ q, const float* __restrict__ k,
    const float* __restrict__ v, const float* __restrict__ beta,
    const float* __restrict__ g, float* __restrict__ o) {
  const int h  = blockIdx.x;
  const int dv = threadIdx.x;            // 0..255
  __shared__ float kq[2][2][DK_N];       // [parity][k/q][dk]

  float hcol[DK_N];
#pragma unroll
  for (int i = 0; i < DK_N; ++i) hcol[i] = 0.f;

  for (int t = 0; t < T_LEN; ++t) {
    const int p = t & 1;
    const size_t row = (size_t)t * H_N + h;
    if (dv < DK_N) kq[p][0][dv]          = k[row * DK_N + dv];
    else           kq[p][1][dv - DK_N]   = q[row * DK_N + (dv - DK_N)];
    __syncthreads();

    const float bt = beta[row];
    const float eg = expf(g[row]);
    const float vt = v[row * DV_N + dv];

    float kh = 0.f;
#pragma unroll
    for (int dk = 0; dk < DK_N; ++dk) kh += kq[p][0][dk] * hcol[dk];
    const float vadj = (vt - kh * eg) * bt;

    float ot = 0.f;
#pragma unroll
    for (int dk = 0; dk < DK_N; ++dk) {
      float hn = hcol[dk] * eg + kq[p][0][dk] * vadj;
      hcol[dk] = hn;
      ot += kq[p][1][dk] * hn;
    }
    o[row * DV_N + dv] = ot;
  }
}

// per-(t,h) RMSNorm of o over DV, * o_norm_scale * silu(gate) -> f16 for final GEMM
__global__ __launch_bounds__(256) void gate_norm_kernel(
    const float* __restrict__ o, const float* __restrict__ gpre,
    const float* __restrict__ oscale, _Float16* __restrict__ out16) {
  const int bid = blockIdx.x;            // t*H + h
  const int dv = threadIdx.x;
  const float val = o[(size_t)bid * DV_N + dv];
  __shared__ float red[256];
  red[dv] = val * val; __syncthreads();
  for (int st = 128; st > 0; st >>= 1) { if (dv < st) red[dv] += red[dv + st]; __syncthreads(); }
  const float rn = rsqrtf(red[0] / (float)DV_N + 1e-5f);
  const float gt = gpre[(size_t)bid * DV_N + dv];
  const float res = val * rn * oscale[dv] * (gt / (1.f + expf(-gt)));
  out16[(size_t)bid * DV_N + dv] = (_Float16)res;
}

// residual add + final RMSNorm
__global__ __launch_bounds__(256) void residual_final_norm(
    const float* __restrict__ x, const float* __restrict__ attn,
    const float* __restrict__ w, float* __restrict__ out) {
  const int t = blockIdx.x, tid = threadIdx.x;
  __shared__ float red[256];
  float yv[8];
  float s = 0.f;
#pragma unroll
  for (int j = 0; j < 8; ++j) {
    int i = tid + 256 * j;
    float v = x[(size_t)t * D_DIM + i] + attn[(size_t)t * D_DIM + i];
    yv[j] = v; s += v * v;
  }
  red[tid] = s; __syncthreads();
  for (int st = 128; st > 0; st >>= 1) { if (tid < st) red[tid] += red[tid + st]; __syncthreads(); }
  const float rn = rsqrtf(red[0] / (float)D_DIM + 1e-5f);
#pragma unroll
  for (int j = 0; j < 8; ++j) {
    int i = tid + 256 * j;
    out[(size_t)t * D_DIM + i] = yv[j] * rn * w[i];
  }
}

// ---------------------------------------------------------------------------
extern "C" void kernel_launch(void* const* d_in, const int* in_sizes, int n_in,
                              void* d_out, int out_size, void* d_ws, size_t ws_size,
                              hipStream_t stream) {
  const float* x          = (const float*)d_in[0];
  const float* norm_scale = (const float*)d_in[1];
  const float* final_ns   = (const float*)d_in[2];
  const float* wq         = (const float*)d_in[3];
  const float* wk         = (const float*)d_in[4];
  const float* wv         = (const float*)d_in[5];
  const float* qconv_w    = (const float*)d_in[6];
  const float* kconv_w    = (const float*)d_in[7];
  const float* vconv_w    = (const float*)d_in[8];
  const float* a_proj     = (const float*)d_in[9];
  const float* A_log      = (const float*)d_in[10];
  const float* dt_bias    = (const float*)d_in[11];
  const float* b_proj     = (const float*)d_in[12];
  const float* g_proj     = (const float*)d_in[13];
  const float* o_ns       = (const float*)d_in[14];
  const float* wo         = (const float*)d_in[15];
  float* out = (float*)d_out;

  char* ws = (char*)d_ws;
  const size_t MB = (size_t)1 << 20;
  float*    xn    = (float*)   (ws + 0   * MB);   // 32 MB  [T,D]
  _Float16* xn16  = (_Float16*)(ws + 32  * MB);   // 16 MB
  _Float16* wbuf  = (_Float16*)(ws + 48  * MB);   // 16 MB (reused per GEMM)
  float*    qpre  = (float*)   (ws + 64  * MB);   // 32 MB
  float*    kpre  = (float*)   (ws + 96  * MB);   // 32 MB
  float*    vpre  = (float*)   (ws + 128 * MB);   // 64 MB
  float*    gate  = (float*)   (ws + 192 * MB);   // 64 MB
  float*    qc    = (float*)   (ws + 256 * MB);   // 32 MB
  float*    kc    = (float*)   (ws + 288 * MB);   // 32 MB
  float*    vc    = (float*)   (ws + 320 * MB);   // 64 MB
  float*    beta  = (float*)   (ws + 384 * MB);   // 256 KB
  float*    gval  = (float*)   (ws + 384 * MB + 262144);
  float*    o_scn = (float*)   (ws + 64  * MB);   // reuse qpre+kpre (64 MB)
  _Float16* o16   = (_Float16*)(ws + 128 * MB);   // reuse vpre[0:32MB]
  float*    attn  = (float*)   (ws + 160 * MB);   // reuse vpre[32:64MB]

  const dim3 blk256(256), blk128(128);

  // 1) RMSNorm + f16 cast of activations
  rmsnorm_cast_kernel<<<T_LEN, blk256, 0, stream>>>(x, norm_scale, xn, xn16);

  // 2-5) Projections via WMMA GEMMs (weights converted to f16 in wbuf)
  cvt_f32_f16<<<(KEY_DIM * D_DIM) / 256, blk256, 0, stream>>>(wq, wbuf, KEY_DIM * D_DIM);
  gemm_f16_wmma<<<dim3(KEY_DIM / 256, T_LEN / 128), blk256, 0, stream>>>(xn16, wbuf, qpre, KEY_DIM, D_DIM);

  cvt_f32_f16<<<(KEY_DIM * D_DIM) / 256, blk256, 0, stream>>>(wk, wbuf, KEY_DIM * D_DIM);
  gemm_f16_wmma<<<dim3(KEY_DIM / 256, T_LEN / 128), blk256, 0, stream>>>(xn16, wbuf, kpre, KEY_DIM, D_DIM);

  cvt_f32_f16<<<(VAL_DIM * D_DIM) / 256, blk256, 0, stream>>>(wv, wbuf, VAL_DIM * D_DIM);
  gemm_f16_wmma<<<dim3(VAL_DIM / 256, T_LEN / 128), blk256, 0, stream>>>(xn16, wbuf, vpre, VAL_DIM, D_DIM);

  cvt_f32_f16<<<(VAL_DIM * D_DIM) / 256, blk256, 0, stream>>>(g_proj, wbuf, VAL_DIM * D_DIM);
  gemm_f16_wmma<<<dim3(VAL_DIM / 256, T_LEN / 128), blk256, 0, stream>>>(xn16, wbuf, gate, VAL_DIM, D_DIM);

  // 6-8) causal conv + silu (+ l2norm for q/k; q folded with DK^-0.5)
  conv_norm_qk<<<T_LEN * H_N, blk128, 0, stream>>>(qpre, qconv_w, qc, 0.08838834764831845f);
  conv_norm_qk<<<T_LEN * H_N, blk128, 0, stream>>>(kpre, kconv_w, kc, 1.0f);
  conv_silu_v<<<(T_LEN * VAL_DIM) / 256, blk256, 0, stream>>>(vpre, vconv_w, vc);

  // 9) beta / log-decay g
  betag_kernel<<<T_LEN * H_N, blk128, 0, stream>>>(xn, a_proj, b_proj, A_log, dt_bias, beta, gval);

  // 10) register-resident gated delta scan (one block per head)
  delta_scan_kernel<<<H_N, blk256, 0, stream>>>(qc, kc, vc, beta, gval, o_scn);

  // 11) output RMSNorm * silu(gate) -> f16
  gate_norm_kernel<<<T_LEN * H_N, blk256, 0, stream>>>(o_scn, gate, o_ns, o16);

  // 12) final projection via WMMA
  cvt_f32_f16<<<(D_DIM * VAL_DIM) / 256, blk256, 0, stream>>>(wo, wbuf, D_DIM * VAL_DIM);
  gemm_f16_wmma<<<dim3(D_DIM / 256, T_LEN / 128), blk256, 0, stream>>>(o16, wbuf, attn, D_DIM, VAL_DIM);

  // 13) residual + final RMSNorm
  residual_final_norm<<<T_LEN, blk256, 0, stream>>>(x, attn, final_ns, out);
}
